// DeepFeatureMatcher_73057393705576
// MI455X (gfx1250) — compile-verified, matrix operand order checked by reading, not run on hardware
//
#include <hip/hip_runtime.h>

// ---------------------------------------------------------------------------
// DeepFeatureMatcher for MI455X (gfx1250, wave32, WMMA, async-LDS copies).
//
// Roofline: sim = d1 @ d2^T is 9216 x 9216 x 512 = 87 GFLOP; everything else
// is < 0.5% of that. We run it as a bf16 WMMA GEMM (f32 accum) and FUSE the
// mutual-NN top-2 reduction into the GEMM epilogue so the 340 MB sim matrix
// is never written to HBM. bf16 descriptors (2 x 9.4 MB) and both activation
// tensors (150 MB) are L2-resident (192 MB), so the GEMM is WMMA-bound and
// the refinement gathers hit L2.
//
// Staging uses GLOBAL_LOAD_ASYNC_TO_LDS_B128 (ASYNCcnt) with explicit
// double-buffering: tile k+1 streams into the alternate LDS buffer while
// WMMAs consume tile k; s_wait_asynccnt leaves the next tile in flight.
// ---------------------------------------------------------------------------

typedef unsigned short u16;
typedef unsigned int   u32;
typedef __attribute__((ext_vector_type(16))) __bf16 v16bf;
typedef __attribute__((ext_vector_type(8)))  float  v8f;
typedef __attribute__((ext_vector_type(4)))  u32    u32x4;
typedef __attribute__((ext_vector_type(4)))  int    v4i;

#define NPIX 9216          // 96*96 descriptors per image
#define KDIM 512           // channels
#define WA   96            // map width/height
#define NTIL 72            // 9216 / 128 tiles per dimension
#define KT   32            // K step per WMMA
#define NKT  (KDIM / KT)   // 16 K-tiles
#define LDT  40            // LDS row stride in halfs (32 + 8 pad -> bank skew)
#define TILE_BYTES (128 * LDT * 2)          // 10,240 B per A/B tile buffer
#define SIM_OFF    (4 * TILE_BYTES)         // 40,960
#define SMEM_BYTES (SIM_OFF + 64 * 129 * 4) // + 33,024 = 73,984 B dynamic LDS
#define EPSF 1e-8f

#if defined(__has_builtin)
#if __has_builtin(__builtin_amdgcn_global_load_async_to_lds_b128)
#define ASYNC_LDS 1
#endif
#endif
#ifndef ASYNC_LDS
#define ASYNC_LDS 0
#endif

#if ASYNC_LDS
// Builtin signature (from hipcc diagnostic): b128 payload is an int4 vector;
// arg0 is addrspace(1) (printed "__device__"), arg1 is LDS addrspace(3).
typedef __attribute__((address_space(1))) v4i gv4i;
typedef __attribute__((address_space(3))) v4i lv4i;
__device__ __forceinline__ void async_b128(const u16* g, u16* l) {
  __builtin_amdgcn_global_load_async_to_lds_b128((gv4i*)g, (lv4i*)l, 0, 0);
}
#if __has_builtin(__builtin_amdgcn_s_wait_asynccnt)
#define WAIT_ASYNC(n) __builtin_amdgcn_s_wait_asynccnt(n)
#else
#define WAIT_ASYNC(n) asm volatile("s_wait_asynccnt %0" ::"i"(n) : "memory")
#endif
#endif

struct T2 { float s0, s1; int i0, i1; };

__device__ inline void t2ins(T2& t, float s, int i) {
  if (s > t.s0)      { t.s1 = t.s0; t.i1 = t.i0; t.s0 = s; t.i0 = i; }
  else if (s > t.s1) { t.s1 = s;    t.i1 = i; }
}

__device__ inline u16 f2bf(float f) {       // round-to-nearest-even f32->bf16
  u32 u = __float_as_uint(f);
  u = (u + 0x7FFFu + ((u >> 16) & 1u)) >> 16;
  return (u16)u;
}

// --------------------------------------------------------------------------
// 1) L2-normalize each descriptor (pixel across 512 channels) and store as
//    bf16, row-major [NPIX][KDIM] so the GEMM stage loads are contiguous.
//    map is [KDIM][NPIX] so both passes are lane-coalesced; the 18.9 MB map
//    sits in L2 for the second pass.
// --------------------------------------------------------------------------
__global__ void __launch_bounds__(256)
normalize_kernel(const float* __restrict__ map, u16* __restrict__ d) {
  int n = blockIdx.x * 256 + threadIdx.x;
  float ss = 0.f;
  for (int k = 0; k < KDIM; ++k) {
    float v = map[(size_t)k * NPIX + n];
    ss += v * v;
  }
  float rn = rsqrtf(ss);
  for (int k0 = 0; k0 < KDIM; k0 += 8) {
    u16 buf[8];
    #pragma unroll
    for (int j = 0; j < 8; ++j)
      buf[j] = f2bf(map[(size_t)(k0 + j) * NPIX + n] * rn);
    *(u32x4*)(d + (size_t)n * KDIM + k0) = *(u32x4*)buf;
  }
}

// --------------------------------------------------------------------------
// 2) Fused bf16 WMMA GEMM + per-tile top-2.
//    Block tile 128x128, 8 waves (4 along M x 2 along N), each wave 32x64 =
//    8 x v_wmma_f32_16x16x32_bf16 per K step. Double-buffered async staging.
//    Epilogue streams the C tile through LDS in two 64-row halves and
//    extracts row/col top-2 partials.
// --------------------------------------------------------------------------
union Frag { u32x4 q[2]; v16bf v; };

__global__ void __launch_bounds__(256)
gemm_top2_kernel(const u16* __restrict__ dn1, const u16* __restrict__ dn2,
                 T2* __restrict__ rowPart, T2* __restrict__ colPart) {
  extern __shared__ __align__(16) char smem[];
  u16* At0 = (u16*)(smem);
  u16* Bt0 = (u16*)(smem + TILE_BYTES);
  u16* At1 = (u16*)(smem + 2 * TILE_BYTES);
  u16* Bt1 = (u16*)(smem + 3 * TILE_BYTES);
  float* simLDS = (float*)(smem + SIM_OFF);   // [64][129] (129 = bank skew)

  const int bx = blockIdx.x;   // column tile (image B rows)
  const int by = blockIdx.y;   // row tile    (image A rows)
  const int tid  = threadIdx.x;
  const int lane = tid & 31, wave = tid >> 5;
  const int wm = wave & 3, wn = wave >> 2;

  v8f acc[2][4];
  #pragma unroll
  for (int mt = 0; mt < 2; ++mt)
    #pragma unroll
    for (int nt = 0; nt < 4; ++nt)
      acc[mt][nt] = 0.f;

  const int r0 = tid >> 2, q = tid & 3;       // staging map: 64 rows x 4 u32x4
  const int half8 = (lane & 16) ? 8 : 0;      // K-half select per A/B layout

  // Per-tile staging: each thread moves 4 x 16 B (2 rows x {A,B}).
  auto issue_tile = [&](u16* At, u16* Bt, int k0) {
    #pragma unroll
    for (int rr = 0; rr < 2; ++rr) {
      int r = r0 + rr * 64;
      const u16* ga = dn1 + (size_t)(by * 128 + r) * KDIM + k0 + q * 8;
      const u16* gb = dn2 + (size_t)(bx * 128 + r) * KDIM + k0 + q * 8;
#if ASYNC_LDS
      async_b128(ga, At + r * LDT + q * 8);
      async_b128(gb, Bt + r * LDT + q * 8);
#else
      *(u32x4*)(At + r * LDT + q * 8) = *(const u32x4*)ga;
      *(u32x4*)(Bt + r * LDT + q * 8) = *(const u32x4*)gb;
#endif
    }
  };

  issue_tile(At0, Bt0, 0);                    // prologue: tile 0 in flight

  for (int kt = 0; kt < NKT; ++kt) {
    u16* Ac = (kt & 1) ? At1 : At0;
    u16* Bc = (kt & 1) ? Bt1 : Bt0;
    // Stream next tile into the alternate buffer while we compute this one.
    if (kt + 1 < NKT)
      issue_tile((kt & 1) ? At0 : At1, (kt & 1) ? Bt0 : Bt1, (kt + 1) * KT);
    if (kt + 2 < NKT) {                       // warm GL2 two tiles ahead
      __builtin_prefetch(dn1 + (size_t)(by * 128 + r0) * KDIM + (kt + 2) * KT, 0, 1);
      __builtin_prefetch(dn2 + (size_t)(bx * 128 + r0) * KDIM + (kt + 2) * KT, 0, 1);
    }
#if ASYNC_LDS
    // My 4 copies for tile kt complete (in-order) once <= 4 remain in flight.
    if (kt + 1 < NKT) { WAIT_ASYNC(4); } else { WAIT_ASYNC(0); }
#endif
    __syncthreads();                          // all waves' tile-kt data landed

    // Fragment loads: lanes 0-15 hold K {0..7,16..23}, lanes 16-31 hold
    // K {8..15,24..31} (CDNA5 16-bit A/B VGPR layout) -> 2 x ds_load_b128.
    Frag a[2], b[4];
    #pragma unroll
    for (int mt = 0; mt < 2; ++mt) {
      int m = wm * 32 + mt * 16 + (lane & 15);
      a[mt].q[0] = *(const u32x4*)(Ac + m * LDT + half8);
      a[mt].q[1] = *(const u32x4*)(Ac + m * LDT + 16 + half8);
    }
    #pragma unroll
    for (int nt = 0; nt < 4; ++nt) {
      int n = wn * 64 + nt * 16 + (lane & 15);
      b[nt].q[0] = *(const u32x4*)(Bc + n * LDT + half8);
      b[nt].q[1] = *(const u32x4*)(Bc + n * LDT + 16 + half8);
    }
    #pragma unroll
    for (int mt = 0; mt < 2; ++mt)
      #pragma unroll
      for (int nt = 0; nt < 4; ++nt)
        acc[mt][nt] = __builtin_amdgcn_wmma_f32_16x16x32_bf16(
            false, a[mt].v, false, b[nt].v, (short)0, acc[mt][nt], false, false);

    __syncthreads();   // reads of this buffer done before it is refilled
  }

  // Epilogue: C tile -> LDS (two 64-row halves) -> row/col top-2 partials.
  const int gRow0 = by * 128, gCol0 = bx * 128;
  T2 colT2 = { -1e30f, -1e30f, 0, 0 };
  const int colInTile = lane & 15;
  const int rAdd = (lane >> 4) * 8;   // C layout: lanes 16-31 hold M = 8+r
  #pragma unroll 1
  for (int h = 0; h < 2; ++h) {
    __syncthreads();
    if ((wm >> 1) == h) {
      #pragma unroll
      for (int mt = 0; mt < 2; ++mt)
        #pragma unroll
        for (int nt = 0; nt < 4; ++nt) {
          int col  = wn * 64 + nt * 16 + colInTile;
          int rowL = (wm & 1) * 32 + mt * 16 + rAdd;
          #pragma unroll
          for (int r = 0; r < 8; ++r)
            simLDS[(rowL + r) * 129 + col] = acc[mt][nt][r];
        }
    }
    __syncthreads();
    if (tid < 64) {                       // row top-2 over this tile's cols
      T2 t = { -1e30f, -1e30f, 0, 0 };
      for (int c = 0; c < 128; ++c)
        t2ins(t, simLDS[tid * 129 + c], gCol0 + c);
      rowPart[(size_t)(gRow0 + h * 64 + tid) * NTIL + bx] = t;
    }
    if (tid < 128) {                      // col top-2 accumulated across halves
      for (int r = 0; r < 64; ++r)
        t2ins(colT2, simLDS[r * 129 + tid], gRow0 + h * 64 + r);
    }
  }
  if (tid < 128)
    colPart[(size_t)(gCol0 + tid) * NTIL + by] = colT2;
}

// --------------------------------------------------------------------------
// 3) Merge 72 per-tile top-2 partials per row/col -> nn index + ratio.
// --------------------------------------------------------------------------
__global__ void __launch_bounds__(256)
reduce_top2_kernel(const T2* __restrict__ part, int* __restrict__ nnOut,
                   float* __restrict__ rOut) {
  int n = blockIdx.x * 256 + threadIdx.x;
  T2 t = { -1e30f, -1e30f, 0, 0 };
  for (int i = 0; i < NTIL; ++i) {
    T2 a = part[(size_t)n * NTIL + i];
    t2ins(t, a.s0, a.i0);
    t2ins(t, a.s1, a.i1);
  }
  float d0 = 2.f - 2.f * t.s0;
  float d1 = 2.f - 2.f * t.s1;
  nnOut[n] = t.i0;
  rOut[n]  = d0 / (d1 + EPSF);
}

// --------------------------------------------------------------------------
// 4) Mutual-NN + ratio + border mask.
// --------------------------------------------------------------------------
__global__ void __launch_bounds__(256)
valid_kernel(const int* __restrict__ nn12, const float* __restrict__ r12,
             const int* __restrict__ nn21, const float* __restrict__ r21,
             int* __restrict__ validOut) {
  int n = blockIdx.x * 256 + threadIdx.x;
  int nn = nn12[n];
  bool ok = (nn21[nn] == n) && (r12[n] <= 0.95f) && (r21[nn] <= 0.95f);
  int xA = n % WA, yA = n / WA, xB = nn % WA, yB = nn / WA;
  bool border = (xA == 0) || (xA == WA - 1) || (yA == 0) || (yA == WA - 1) ||
                (xB == 0) || (xB == WA - 1) || (yB == 0) || (yB == WA - 1);
  validOut[n] = (ok && !border) ? 1 : 0;
}

// --------------------------------------------------------------------------
// 5) Refinement: one wave32 per point. Lanes stride channels; 4x4 score
//    matrix (with fused per-pixel act normalization) via shfl_xor trees.
//    Writes all four outputs (refined_A, refined_B, mask, scores) as floats.
// --------------------------------------------------------------------------
__global__ void __launch_bounds__(256)
refine_kernel(const float* __restrict__ actA, const float* __restrict__ actB,
              const int* __restrict__ valid, const int* __restrict__ nn12,
              float* __restrict__ out) {
  const int W = 192, HW = 192 * 192;
  int wave = threadIdx.x >> 5, lane = threadIdx.x & 31;
  int p = blockIdx.x * 8 + wave;
  int v  = valid[p];
  int nn = nn12[p];
  int pAx = v ? (p % WA)  : 0, pAy = v ? (p / WA)  : 0;
  int pBx = v ? (nn % WA) : 0, pBy = v ? (nn / WA) : 0;
  const int nbx[4] = {0, 0, 1, 1};
  const int nby[4] = {0, 1, 0, 1};
  int aOff[4], bOff[4];
  #pragma unroll
  for (int i = 0; i < 4; ++i) {
    aOff[i] = (2 * pAy + nby[i]) * W + (2 * pAx + nbx[i]);
    bOff[i] = (2 * pBy + nby[i]) * W + (2 * pBx + nbx[i]);
  }

  float dot[4][4] = {}, na[4] = {}, nb[4] = {};
  for (int c = lane; c < KDIM; c += 32) {
    size_t base = (size_t)c * HW;
    float av[4], bv[4];
    #pragma unroll
    for (int i = 0; i < 4; ++i) { av[i] = actA[base + aOff[i]]; bv[i] = actB[base + bOff[i]]; }
    #pragma unroll
    for (int i = 0; i < 4; ++i) {
      na[i] += av[i] * av[i];
      nb[i] += bv[i] * bv[i];
      #pragma unroll
      for (int j = 0; j < 4; ++j) dot[i][j] += av[i] * bv[j];
    }
  }
  #pragma unroll
  for (int o = 16; o > 0; o >>= 1) {
    #pragma unroll
    for (int i = 0; i < 4; ++i) {
      na[i] += __shfl_xor(na[i], o);
      nb[i] += __shfl_xor(nb[i], o);
      #pragma unroll
      for (int j = 0; j < 4; ++j) dot[i][j] += __shfl_xor(dot[i][j], o);
    }
  }

  float s[4][4], rna[4], rnb[4];
  #pragma unroll
  for (int i = 0; i < 4; ++i) { rna[i] = rsqrtf(na[i]); rnb[i] = rsqrtf(nb[i]); }
  #pragma unroll
  for (int i = 0; i < 4; ++i)
    #pragma unroll
    for (int j = 0; j < 4; ++j) s[i][j] = dot[i][j] * rna[i] * rnb[j];

  int mA[4], mB[4]; float ratA[4], ratB[4], scA[4];
  #pragma unroll
  for (int i = 0; i < 4; ++i) {          // A->B: top-2 per row
    float t0 = -1e30f, t1 = -1e30f; int j0 = 0;
    #pragma unroll
    for (int j = 0; j < 4; ++j) {
      if (s[i][j] > t0) { t1 = t0; t0 = s[i][j]; j0 = j; }
      else if (s[i][j] > t1) t1 = s[i][j];
    }
    float d0 = 2.f - 2.f * t0, d1 = 2.f - 2.f * t1;
    ratA[i] = d0 / (d1 + EPSF); scA[i] = d0; mA[i] = j0;
  }
  #pragma unroll
  for (int j = 0; j < 4; ++j) {          // B->A: top-2 per column
    float t0 = -1e30f, t1 = -1e30f; int i0c = 0;
    #pragma unroll
    for (int i = 0; i < 4; ++i) {
      if (s[i][j] > t0) { t1 = t0; t0 = s[i][j]; i0c = i; }
      else if (s[i][j] > t1) t1 = s[i][j];
    }
    float d0 = 2.f - 2.f * t0, d1 = 2.f - 2.f * t1;
    ratB[j] = d0 / (d1 + EPSF); mB[j] = i0c;
  }

  bool mk[4];
  #pragma unroll
  for (int i = 0; i < 4; ++i)
    mk[i] = (fminf(ratA[i], ratB[i]) < 0.9f) && (mB[mA[i]] == i);

  // Discard the two largest masked scores (reference's top_k(score_masked,2)).
  float sm[4];
  #pragma unroll
  for (int i = 0; i < 4; ++i) sm[i] = mk[i] ? scA[i] : 5.0f;
  int d0i = 0; float b0 = sm[0];
  #pragma unroll
  for (int t = 1; t < 4; ++t) if (sm[t] > b0) { b0 = sm[t]; d0i = t; }
  int d1i = 0; float b1 = -1e30f;
  #pragma unroll
  for (int t = 0; t < 4; ++t)
    if (t != d0i && sm[t] > b1) { b1 = sm[t]; d1i = t; }
  mk[d0i] = false; mk[d1i] = false;
  #pragma unroll
  for (int i = 0; i < 4; ++i) mk[i] = mk[i] && (v != 0);

  if (lane == 0) {
    const int P4 = NPIX * 4;
    float* rA = out;                 // refined_A [2][P][4]
    float* rB = out + 2 * P4;        // refined_B [2][P][4]
    float* mO = out + 4 * P4;        // mask      [P][4]
    float* sO = out + 5 * P4;        // scores    [P][4][4]
    #pragma unroll
    for (int i = 0; i < 4; ++i) {
      rA[p * 4 + i]      = (float)(2 * pAx + nbx[i]);
      rA[P4 + p * 4 + i] = (float)(2 * pAy + nby[i]);
      rB[p * 4 + i]      = (float)(2 * pBx + nbx[mA[i]]);
      rB[P4 + p * 4 + i] = (float)(2 * pBy + nby[mA[i]]);
      mO[p * 4 + i]      = mk[i] ? 1.f : 0.f;
      #pragma unroll
      for (int j = 0; j < 4; ++j) sO[p * 16 + i * 4 + j] = s[i][j];
    }
  }
}

// --------------------------------------------------------------------------
extern "C" void kernel_launch(void* const* d_in, const int* in_sizes, int n_in,
                              void* d_out, int out_size, void* d_ws, size_t ws_size,
                              hipStream_t stream) {
  const float* mapA = (const float*)d_in[0];
  const float* mapB = (const float*)d_in[1];
  const float* actA = (const float*)d_in[2];
  const float* actB = (const float*)d_in[3];
  float* out = (float*)d_out;

  char* ws = (char*)d_ws;
  size_t off = 0;
  auto alloc = [&](size_t sz) -> void* {
    void* p = ws + off;
    off = (off + sz + 255) & ~(size_t)255;
    return p;
  };
  u16* dn1     = (u16*)alloc((size_t)NPIX * KDIM * 2);         //  9.4 MB
  u16* dn2     = (u16*)alloc((size_t)NPIX * KDIM * 2);         //  9.4 MB
  T2*  rowPart = (T2*)alloc((size_t)NPIX * NTIL * sizeof(T2)); // 10.6 MB
  T2*  colPart = (T2*)alloc((size_t)NPIX * NTIL * sizeof(T2)); // 10.6 MB
  int*   nn12  = (int*)alloc((size_t)NPIX * 4);
  float* r12   = (float*)alloc((size_t)NPIX * 4);
  int*   nn21  = (int*)alloc((size_t)NPIX * 4);
  float* r21   = (float*)alloc((size_t)NPIX * 4);
  int*   validM= (int*)alloc((size_t)NPIX * 4);

  normalize_kernel<<<NPIX / 256, 256, 0, stream>>>(mapA, dn1);
  normalize_kernel<<<NPIX / 256, 256, 0, stream>>>(mapB, dn2);
  gemm_top2_kernel<<<dim3(NTIL, NTIL), 256, SMEM_BYTES, stream>>>(dn1, dn2, rowPart, colPart);
  reduce_top2_kernel<<<NPIX / 256, 256, 0, stream>>>(rowPart, nn12, r12);
  reduce_top2_kernel<<<NPIX / 256, 256, 0, stream>>>(colPart, nn21, r21);
  valid_kernel<<<NPIX / 256, 256, 0, stream>>>(nn12, r12, nn21, r21, validM);
  refine_kernel<<<NPIX / 8, 256, 0, stream>>>(actA, actB, validM, nn12, out);
}
